// TemporalPyGGraphLayer_14053132993208
// MI455X (gfx1250) — matrix operand order repaired and checked by reading, not compiled, not run on hardware
//
#include <hip/hip_runtime.h>
#include <hip/hip_bf16.h>
#include <math.h>

typedef _Float16 v16h __attribute__((ext_vector_type(16)));
typedef _Float16 v8h  __attribute__((ext_vector_type(8)));
typedef float    v8f  __attribute__((ext_vector_type(8)));
typedef float    v4f  __attribute__((ext_vector_type(4)));

#define GAT_B 4
#define GAT_T 4096
#define GAT_N (GAT_B * GAT_T)      // 16384
#define GAT_D 128
#define GAT_H 4
#define GAT_O 32
#define NEG_SLOPE 0.2f
#define GAT_EPS 1e-16f

__device__ __forceinline__ v16h cat16(v8h lo, v8h hi) {
    return __builtin_shufflevector(lo, hi, 0,1,2,3,4,5,6,7,8,9,10,11,12,13,14,15);
}

// ---------------------------------------------------------------------------
// Kernel 1: h = x @ W^T  via v_wmma_f32_16x16x32_f16
// One workgroup (256 thr = 8 waves) computes 128 rows; each wave a 16x128 strip.
// A (x tile, 16x32 f16) converted from global f32 into ISA lane layout.
// B (W, row-major [n,k] == B^T) staged f16 in LDS, fragments via ds_load_b128.
// ---------------------------------------------------------------------------
__global__ __launch_bounds__(256) void k_gemm(const float* __restrict__ x,
                                              const float* __restrict__ W,
                                              float* __restrict__ hout) {
    __shared__ _Float16 sW[GAT_D * GAT_D];   // 32 KB

    const int tid = threadIdx.x;
    // Stage W (f32 -> f16), 8 elems per iteration per thread
    for (int i = tid; i < GAT_D * GAT_D / 8; i += 256) {
        v4f w0 = ((const v4f*)W)[2 * i + 0];
        v4f w1 = ((const v4f*)W)[2 * i + 1];
        v8h hv;
        hv[0] = (_Float16)w0[0]; hv[1] = (_Float16)w0[1];
        hv[2] = (_Float16)w0[2]; hv[3] = (_Float16)w0[3];
        hv[4] = (_Float16)w1[0]; hv[5] = (_Float16)w1[1];
        hv[6] = (_Float16)w1[2]; hv[7] = (_Float16)w1[3];
        ((v8h*)sW)[i] = hv;
    }
    __syncthreads();

    const int wave = tid >> 5;
    const int lane = tid & 31;
    const int lhi  = lane >> 4;      // half-wave select
    const int llo  = lane & 15;

    // A fragments: row = this lane's M row; K layout per ISA:
    // lanes 0-15: K in [kb, kb+8) U [kb+16, kb+24) with kb = ks*32 (+8 for hi half-wave)
    const size_t rowA = (size_t)blockIdx.x * 128 + wave * 16 + llo;
    const float* xr = x + rowA * GAT_D;
    v16h afr[4];
#pragma unroll
    for (int ks = 0; ks < 4; ++ks) {
        const int kb = ks * 32 + lhi * 8;
        v4f f0 = *(const v4f*)(xr + kb);
        v4f f1 = *(const v4f*)(xr + kb + 4);
        v4f f2 = *(const v4f*)(xr + kb + 16);
        v4f f3 = *(const v4f*)(xr + kb + 20);
        v16h a;
#pragma unroll
        for (int j = 0; j < 4; ++j) {
            a[j]      = (_Float16)f0[j];
            a[4 + j]  = (_Float16)f1[j];
            a[8 + j]  = (_Float16)f2[j];
            a[12 + j] = (_Float16)f3[j];
        }
        afr[ks] = a;
    }

    const size_t rowDbase = (size_t)blockIdx.x * 128 + wave * 16 + lhi * 8;
#pragma unroll
    for (int t = 0; t < 8; ++t) {             // 8 N-tiles of 16 cols
        v8f acc = {0.f,0.f,0.f,0.f,0.f,0.f,0.f,0.f};
#pragma unroll
        for (int ks = 0; ks < 4; ++ks) {
            // B fragment: lane holds column n = llo; K in [kbB, kbB+8) U [kbB+16, kbB+24),
            // kbB = ks*32 (+8 for hi half-wave), matching A.
            const int rB  = t * 16 + llo;
            const int kbB = ks * 32 + lhi * 8;
            const v8h* pb = (const v8h*)&sW[rB * GAT_D + kbB];
            v16h b = cat16(pb[0], pb[2]);
            acc = __builtin_amdgcn_wmma_f32_16x16x32_f16(
                false, afr[ks], false, b, (short)0, acc, false, false);
        }
#pragma unroll
        for (int r = 0; r < 8; ++r)
            hout[(rowDbase + r) * GAT_D + t * 16 + llo] = acc[r];
    }
}

// ---------------------------------------------------------------------------
// Kernel: attention projections  a_src/a_dst [N,H]
// (n*H + hh)*O == n*128 + hh*32, so idx indexes h directly.
// ---------------------------------------------------------------------------
__global__ __launch_bounds__(256) void k_att(const float* __restrict__ h,
                                             const float* __restrict__ att_src,
                                             const float* __restrict__ att_dst,
                                             float* __restrict__ a_s,
                                             float* __restrict__ a_d, int NH) {
    int idx = blockIdx.x * 256 + threadIdx.x;
    if (idx >= NH) return;
    const int hh = idx & (GAT_H - 1);
    const float* hp = h + (size_t)idx * GAT_O;
    const float* As = att_src + hh * GAT_O;
    const float* Ad = att_dst + hh * GAT_O;
    float s = 0.f, d = 0.f;
#pragma unroll
    for (int o = 0; o < GAT_O; ++o) {
        float v = hp[o];
        s += v * As[o];
        d += v * Ad[o];
    }
    a_s[idx] = s;
    a_d[idx] = d;
}

// ---------------------------------------------------------------------------
// Init: out = bias broadcast; m = -inf; s = 0
// ---------------------------------------------------------------------------
__global__ __launch_bounds__(256) void k_init(float* __restrict__ out,
                                              const float* __restrict__ bias,
                                              float* __restrict__ m,
                                              float* __restrict__ s,
                                              int total, int NH) {
    int idx = blockIdx.x * 256 + threadIdx.x;
    if (idx < total) out[idx] = bias[idx & (GAT_D - 1)];
    if (idx < NH) { m[idx] = __int_as_float(0xFF800000); s[idx] = 0.f; }
}

// ---------------------------------------------------------------------------
// Edge decode: batched temporal edges + self loops (computed on the fly)
// ---------------------------------------------------------------------------
__device__ __forceinline__ void decode_edge(unsigned ge, const long long* __restrict__ ei,
                                            unsigned E, unsigned BE, int& s, int& d) {
    if (ge < BE) {
        unsigned b = ge / E;
        unsigned e = ge - b * E;
        int off = (int)b * GAT_T;
        s = (int)ei[e] + off;
        d = (int)ei[E + e] + off;
    } else {
        s = d = (int)(ge - BE);
    }
}

__device__ __forceinline__ void atomicMaxFloat(float* addr, float val) {
    if (val >= 0.f) atomicMax((int*)addr, __float_as_int(val));
    else            atomicMin((unsigned int*)addr, __float_as_uint(val));
}

// Pass A: segment max of leaky-relu logits, one thread per (edge, head)
__global__ __launch_bounds__(256) void k_edge_max(const long long* __restrict__ ei,
                                                  const float* __restrict__ a_s,
                                                  const float* __restrict__ a_d,
                                                  float* __restrict__ m,
                                                  unsigned E, unsigned BE, unsigned totH) {
    unsigned idx = blockIdx.x * 256 + threadIdx.x;
    if (idx >= totH) return;
    unsigned ge = idx >> 2;
    int hh = idx & 3;
    int src, dst;
    decode_edge(ge, ei, E, BE, src, dst);
    float e = a_s[src * GAT_H + hh] + a_d[dst * GAT_H + hh];
    e = (e > 0.f) ? e : NEG_SLOPE * e;
    atomicMaxFloat(&m[dst * GAT_H + hh], e);
}

// Pass B: exp(e - max), cache per-edge exp, segment sum
__global__ __launch_bounds__(256) void k_edge_exp(const long long* __restrict__ ei,
                                                  const float* __restrict__ a_s,
                                                  const float* __restrict__ a_d,
                                                  const float* __restrict__ m,
                                                  float* __restrict__ ssum,
                                                  float* __restrict__ ex,
                                                  unsigned E, unsigned BE, unsigned totH) {
    unsigned idx = blockIdx.x * 256 + threadIdx.x;
    if (idx >= totH) return;
    unsigned ge = idx >> 2;
    int hh = idx & 3;
    int src, dst;
    decode_edge(ge, ei, E, BE, src, dst);
    float e = a_s[src * GAT_H + hh] + a_d[dst * GAT_H + hh];
    e = (e > 0.f) ? e : NEG_SLOPE * e;
    float xv = __expf(e - m[dst * GAT_H + hh]);
    ex[idx] = xv;
    atomicAdd(&ssum[dst * GAT_H + hh], xv);
}

// Pass C: scatter h[src]*alpha into out[dst]; one wave per edge, 4 channels/lane.
__global__ __launch_bounds__(256) void k_scatter(const long long* __restrict__ ei,
                                                 const float* __restrict__ h,
                                                 const float* __restrict__ ex,
                                                 const float* __restrict__ ssum,
                                                 float* __restrict__ out,
                                                 unsigned E, unsigned BE, unsigned Etot) {
    unsigned tid = blockIdx.x * 256 + threadIdx.x;
    unsigned ge = tid >> 5;
    if (ge >= Etot) return;
    const int lane = threadIdx.x & 31;
    const int c0 = lane * 4;
    const int hh = lane >> 3;          // c0 >> 5
    int src, dst;
    decode_edge(ge, ei, E, BE, src, dst);
    v4f hv = *(const v4f*)(h + (size_t)src * GAT_D + c0);
    float alpha = ex[ge * GAT_H + hh] / (ssum[dst * GAT_H + hh] + GAT_EPS);
    float* op = out + (size_t)dst * GAT_D + c0;
    atomicAdd(op + 0, hv[0] * alpha);
    atomicAdd(op + 1, hv[1] * alpha);
    atomicAdd(op + 2, hv[2] * alpha);
    atomicAdd(op + 3, hv[3] * alpha);
}

extern "C" void kernel_launch(void* const* d_in, const int* in_sizes, int n_in,
                              void* d_out, int out_size, void* d_ws, size_t ws_size,
                              hipStream_t stream) {
    const float*     x       = (const float*)d_in[0];
    const long long* ei      = (const long long*)d_in[1];
    const float*     W       = (const float*)d_in[2];
    const float*     att_src = (const float*)d_in[3];
    const float*     att_dst = (const float*)d_in[4];
    const float*     bias    = (const float*)d_in[5];
    float*           out     = (float*)d_out;

    const int N = GAT_N;
    const unsigned E    = (unsigned)(in_sizes[1] / 2);
    const unsigned BE   = (unsigned)GAT_B * E;
    const unsigned Etot = BE + (unsigned)N;

    // Workspace layout (floats)
    float* h  = (float*)d_ws;                 // N*128
    float* as = h  + (size_t)N * GAT_D;       // N*H
    float* ad = as + (size_t)N * GAT_H;       // N*H
    float* mx = ad + (size_t)N * GAT_H;       // N*H  (segment max)
    float* sm = mx + (size_t)N * GAT_H;       // N*H  (segment sum)
    float* ex = sm + (size_t)N * GAT_H;       // Etot*H (per-edge exp)

    const int totalOut = N * GAT_D;
    const int NH = N * GAT_H;
    k_init<<<(totalOut + 255) / 256, 256, 0, stream>>>(out, bias, mx, sm, totalOut, NH);

    k_gemm<<<N / 128, 256, 0, stream>>>(x, W, h);

    k_att<<<(NH + 255) / 256, 256, 0, stream>>>(h, att_src, att_dst, as, ad, NH);

    const unsigned totH = Etot * GAT_H;
    k_edge_max<<<(totH + 255) / 256, 256, 0, stream>>>(ei, as, ad, mx, E, BE, totH);
    k_edge_exp<<<(totH + 255) / 256, 256, 0, stream>>>(ei, as, ad, mx, sm, ex, E, BE, totH);

    const unsigned scatterThreads = Etot * 32u;
    k_scatter<<<(scatterThreads + 255) / 256, 256, 0, stream>>>(ei, h, ex, sm, out, E, BE, Etot);
}
